// MultiHeadSelfAttention_72361609003117
// MI455X (gfx1250) — compile-verified, matrix operand order checked by reading, not run on hardware
//
#include <hip/hip_runtime.h>

// ---------------------------------------------------------------------------
// MHSA for B=2, T=2048, C=1024, H=16, D=64 on gfx1250 (wave32, WMMA bf16).
//   Kernel 1: qkv = x @ W^T + b   (M=4096, K=1024, N=3072), writes bf16
//             Q[BH][T][D], K[BH][T][D], Vt[BH][D][T] into workspace.
//             Global loads software-pipelined under the WMMAs.
//   Kernel 2: flash attention per (head, 128-row query tile); K/Vt tiles
//             double-buffered in LDS via Tensor Data Mover async loads.
// ---------------------------------------------------------------------------

typedef __attribute__((ext_vector_type(16))) __bf16        v16bf;
typedef __attribute__((ext_vector_type(8)))  float         v8f;
typedef __attribute__((ext_vector_type(8)))  unsigned int  v8u;
typedef __attribute__((ext_vector_type(4)))  unsigned int  v4u;
typedef __attribute__((ext_vector_type(4)))  float         v4f;
typedef __attribute__((ext_vector_type(8)))  int           v8i;
typedef __attribute__((ext_vector_type(4)))  int           v4i;

#define LDS_FENCE() asm volatile("s_wait_dscnt 0" ::: "memory")

static __device__ __forceinline__ unsigned short f2bf(float f) {
    unsigned u = __builtin_bit_cast(unsigned, f);
    u = (u + 0x7FFFu + ((u >> 16) & 1u)) >> 16;   // round-to-nearest-even
    return (unsigned short)u;
}
static __device__ __forceinline__ unsigned pack2bf(float lo, float hi) {
    return (unsigned)f2bf(lo) | ((unsigned)f2bf(hi) << 16);
}

// A/B fragment for v_wmma_f32_16x16x32_bf16: per lane two contiguous 16-byte
// chunks (K = grp*8..+7 and K = 16+grp*8..+7), rows striped over lanes 0..15.
static __device__ __forceinline__ v16bf ldfrag(const unsigned short* p0,
                                               const unsigned short* p1) {
    v4u a = *(const v4u*)p0;
    v4u b = *(const v4u*)p1;
    v8u u = __builtin_shufflevector(a, b, 0, 1, 2, 3, 4, 5, 6, 7);
    return __builtin_bit_cast(v16bf, u);
}

static __device__ __forceinline__ v8f wmma_bf16(v16bf a, v16bf b, v8f c) {
    return __builtin_amdgcn_wmma_f32_16x16x32_bf16(
        /*neg_a=*/false, a, /*neg_b=*/false, b,
        /*c_mod=*/(short)0, c, /*reuse_a=*/false, /*reuse_b=*/false);
}

// ---------------------------------------------------------------------------
// TDM: async-load a (tile_h x tile_w) bf16 tile (row stride in elements) into
// LDS, padding each 32 LDS DWORDs (64 bf16) with 4 DWORDs (8 bf16) so rows
// land on a 72-element (144 B) stride — matching the compute-side layout.
// ---------------------------------------------------------------------------
static __device__ __forceinline__ void tdm_load_2d_bf16(
    unsigned lds_addr, const void* gptr, unsigned tile_w, unsigned tile_h,
    unsigned long long row_stride_elems) {
    const unsigned long long ga = (unsigned long long)(uintptr_t)gptr;
    v4u g0;
    g0[0] = 1u;                                       // count=1, user mode
    g0[1] = lds_addr;                                 // LDS byte address
    g0[2] = (unsigned)ga;                             // global_addr[31:0]
    g0[3] = (unsigned)((ga >> 32) & 0x01FFFFFFull)    // global_addr[56:32]
            | (2u << 30);                             // type = 2 (image)
    const unsigned flags = (1u << 16)                 // data_size = 2 bytes
                         | (1u << 20)                 // pad_enable
                         | (4u << 22)                 // pad_interval = 32 dw
                         | (3u << 25);                // pad_amount   = 4 dw
    const unsigned long long td = 1ull << 20;         // tensor dims: no clamp
    const unsigned long long s0 = row_stride_elems;   // tensor_dim0_stride
    v8u g1u;
    g1u[0] = flags;
    g1u[1] = (unsigned)((td & 0xFFFFull) << 16);                  // td0[15:0]
    g1u[2] = (unsigned)((td >> 16) & 0xFFFFull)                   // td0[31:16]
           | (unsigned)((td & 0xFFFFull) << 16);                  // td1[15:0]
    g1u[3] = (unsigned)((td >> 16) & 0xFFFFull) | (tile_w << 16); // td1|tile0
    g1u[4] = tile_h;                                  // tile_dim1, tile_dim2=0
    g1u[5] = (unsigned)s0;
    g1u[6] = (unsigned)((s0 >> 32) & 0xFFFFull);      // stride0 hi | stride1 lo
    g1u[7] = 0u;
    v8i g1 = __builtin_bit_cast(v8i, g1u);
    v4i z4 = {0, 0, 0, 0};
#if __clang_major__ >= 23
    v8i z8 = {0, 0, 0, 0, 0, 0, 0, 0};
    __builtin_amdgcn_tensor_load_to_lds(g0, g1, z4, z4, z8, 0);
#else
    __builtin_amdgcn_tensor_load_to_lds(g0, g1, z4, z4, 0);
#endif
}

// ---------------------------------------------------------------------------
// Kernel 1: QKV projection GEMM.  grid = (24, 32), block = 256 (8 waves)
// ---------------------------------------------------------------------------
#define LDA 40   // 32 bf16 + 8 pad: 80B row stride, 16B aligned, conflict-free

__global__ __launch_bounds__(256) void qkv_gemm_kernel(
    const float* __restrict__ x, const float* __restrict__ W,
    const float* __restrict__ bias,
    unsigned short* __restrict__ qb, unsigned short* __restrict__ kb,
    unsigned short* __restrict__ vtb) {
    __shared__ unsigned short sA[128 * LDA];
    __shared__ unsigned short sB[128 * LDA];

    const int tid  = threadIdx.x;
    const int lane = tid & 31;
    const int wave = tid >> 5;
    const int grp  = lane >> 4;     // 0 | 1 (K-chunk group)
    const int frow = lane & 15;     // fragment row/col index

    const int m0 = blockIdx.y * 128;
    const int n0 = blockIdx.x * 128;
    const int mw = (wave >> 2) * 64;   // wave M offset (0 | 64)
    const int nw = (wave & 3) * 32;    // wave N offset (0,32,64,96)

    v8f acc[4][2];
#pragma unroll
    for (int i = 0; i < 4; ++i)
#pragma unroll
        for (int j = 0; j < 2; ++j)
#pragma unroll
            for (int r = 0; r < 8; ++r) acc[i][j][r] = 0.0f;

    // staging: thread -> (row = tid/2, 16 consecutive f32 at col (tid&1)*16)
    const int lrow = tid >> 1;
    const int lcol = (tid & 1) * 16;
    const float* xrow = x + (m0 + lrow) * 1024 + lcol;
    const float* wrow = W + (n0 + lrow) * 1024 + lcol;
    unsigned short* sAp = sA + lrow * LDA + lcol;
    unsigned short* sBp = sB + lrow * LDA + lcol;

    // Software pipeline: global f32 loads for step k0 are issued during the
    // WMMAs of step k0-32 and only converted/stored at the top of step k0.
    v4f ra[4], rb[4];
#pragma unroll
    for (int q = 0; q < 4; ++q) {
        ra[q] = *(const v4f*)(xrow + q * 4);
        rb[q] = *(const v4f*)(wrow + q * 4);
    }

    for (int k0 = 0; k0 < 1024; k0 += 32) {
        __syncthreads();   // previous tile fully consumed
        {
            v4u pa0 = {pack2bf(ra[0][0], ra[0][1]), pack2bf(ra[0][2], ra[0][3]),
                       pack2bf(ra[1][0], ra[1][1]), pack2bf(ra[1][2], ra[1][3])};
            v4u pa1 = {pack2bf(ra[2][0], ra[2][1]), pack2bf(ra[2][2], ra[2][3]),
                       pack2bf(ra[3][0], ra[3][1]), pack2bf(ra[3][2], ra[3][3])};
            v4u pb0 = {pack2bf(rb[0][0], rb[0][1]), pack2bf(rb[0][2], rb[0][3]),
                       pack2bf(rb[1][0], rb[1][1]), pack2bf(rb[1][2], rb[1][3])};
            v4u pb1 = {pack2bf(rb[2][0], rb[2][1]), pack2bf(rb[2][2], rb[2][3]),
                       pack2bf(rb[3][0], rb[3][1]), pack2bf(rb[3][2], rb[3][3])};
            *(v4u*)(sAp + 0) = pa0;
            *(v4u*)(sAp + 8) = pa1;
            *(v4u*)(sBp + 0) = pb0;
            *(v4u*)(sBp + 8) = pb1;
        }
        __syncthreads();   // tile published

        if (k0 + 32 < 1024) {   // issue next step's loads under the WMMAs
            const int kn = k0 + 32;
#pragma unroll
            for (int q = 0; q < 4; ++q) {
                ra[q] = *(const v4f*)(xrow + kn + q * 4);
                rb[q] = *(const v4f*)(wrow + kn + q * 4);
            }
        }

        v16bf af[4], bf[2];
#pragma unroll
        for (int mi = 0; mi < 4; ++mi) {
            const unsigned short* base = sA + (mw + mi * 16 + frow) * LDA;
            af[mi] = ldfrag(base + grp * 8, base + 16 + grp * 8);
        }
#pragma unroll
        for (int ni = 0; ni < 2; ++ni) {
            const unsigned short* base = sB + (nw + ni * 16 + frow) * LDA;
            bf[ni] = ldfrag(base + grp * 8, base + 16 + grp * 8);
        }
#pragma unroll
        for (int ni = 0; ni < 2; ++ni)
#pragma unroll
            for (int mi = 0; mi < 4; ++mi)
                acc[mi][ni] = wmma_bf16(af[mi], bf[ni], acc[mi][ni]);
    }

    // Epilogue: + bias, bf16, scatter into Q / K / Vt head layouts.
#pragma unroll
    for (int mi = 0; mi < 4; ++mi)
#pragma unroll
        for (int ni = 0; ni < 2; ++ni) {
            const int n = n0 + nw + ni * 16 + frow;
            const float bn = bias[n];
#pragma unroll
            for (int r = 0; r < 8; ++r) {
                const int m = m0 + mw + mi * 16 + r + grp * 8;
                const unsigned short h = f2bf(acc[mi][ni][r] + bn);
                const int b = m >> 11, t = m & 2047;
                if (n < 1024) {
                    const int hh = n >> 6, d = n & 63;
                    qb[((b * 16 + hh) * 2048 + t) * 64 + d] = h;
                } else if (n < 2048) {
                    const int nn = n - 1024, hh = nn >> 6, d = nn & 63;
                    kb[((b * 16 + hh) * 2048 + t) * 64 + d] = h;
                } else {
                    const int nn = n - 2048, hh = nn >> 6, d = nn & 63;
                    vtb[((b * 16 + hh) * 64 + d) * 2048 + t] = h;
                }
            }
        }
}

// ---------------------------------------------------------------------------
// Kernel 2: flash attention.  grid = (BH=32, T/128=16), block = 256 (8 waves)
// Each wave owns a 16-row query tile; 64-key tiles double-buffered in LDS by
// TDM: wave 0 issues tile kt+1 while the block computes tile kt, then waits
// tensorcnt<=2 (TDM ops of one wave retire in order) to publish tile kt.
// ---------------------------------------------------------------------------
#define LDK 72   // 64 bf16 + 8 pad: 144B stride (TDM pad 4dw per 32dw)

__global__ __launch_bounds__(256) void mhsa_flash_kernel(
    const unsigned short* __restrict__ qb, const unsigned short* __restrict__ kb,
    const unsigned short* __restrict__ vtb, float* __restrict__ out) {
    __shared__ unsigned short sK[2][64 * LDK];
    __shared__ unsigned short sV[2][64 * LDK];
    __shared__ unsigned short sP[8 * 16 * LDK];

    const int tid  = threadIdx.x;
    const int lane = tid & 31;
    const int wave = tid >> 5;
    const int grp  = lane >> 4;
    const int frow = lane & 15;
    const int bh   = blockIdx.x;      // b*16 + h
    const int qt   = blockIdx.y;      // query tile (128 rows)

    // Resident Q fragments (16 rows x 64 k, two 32-wide k-steps).
    v16bf qf[2];
    {
        const unsigned short* qp =
            qb + (bh * 2048 + qt * 128 + wave * 16 + frow) * 64;
        qf[0] = ldfrag(qp + grp * 8, qp + 16 + grp * 8);
        qf[1] = ldfrag(qp + 32 + grp * 8, qp + 48 + grp * 8);
    }

    v8f accO[4];
    float mrow[8], lsum[8];
#pragma unroll
    for (int dt = 0; dt < 4; ++dt)
#pragma unroll
        for (int r = 0; r < 8; ++r) accO[dt][r] = 0.0f;
#pragma unroll
    for (int r = 0; r < 8; ++r) { mrow[r] = -1e30f; lsum[r] = 0.0f; }

    // TDM source bases (block-uniform -> SGPR descriptors).
    const unsigned short* kgb = kb + bh * 2048 * 64;   // + kt*64*64
    const unsigned short* vgb = vtb + bh * 64 * 2048;  // + kt*64
    const unsigned sKl[2] = {(unsigned)(uintptr_t)(void*)sK[0],
                             (unsigned)(uintptr_t)(void*)sK[1]};
    const unsigned sVl[2] = {(unsigned)(uintptr_t)(void*)sV[0],
                             (unsigned)(uintptr_t)(void*)sV[1]};
    unsigned short* sPw = sP + wave * 16 * LDK;

    const float scale = 0.125f;  // 1/sqrt(64)

    // Prologue: kick off tile 0.
    if (wave == 0) {
        tdm_load_2d_bf16(sKl[0], kgb, 64, 64, 64ull);
        tdm_load_2d_bf16(sVl[0], vgb, 64, 64, 2048ull);
    }

    for (int kt = 0; kt < 32; ++kt) {
        const int cur = kt & 1;
        __syncthreads();   // buffer cur^1 no longer being read
        if (wave == 0) {
            if (kt + 1 < 32) {
                tdm_load_2d_bf16(sKl[cur ^ 1], kgb + (kt + 1) * 4096,
                                 64, 64, 64ull);
                tdm_load_2d_bf16(sVl[cur ^ 1], vgb + (kt + 1) * 64,
                                 64, 64, 2048ull);
                __builtin_amdgcn_s_wait_tensorcnt(2);  // tile kt landed
            } else {
                __builtin_amdgcn_s_wait_tensorcnt(0);
            }
        }
        __syncthreads();   // tile kt visible to all 8 waves
        const unsigned short* sKc = sK[cur];
        const unsigned short* sVc = sV[cur];

        // ---- S = Q K^T (16 x 64) ----
        v8f s[4];
#pragma unroll
        for (int si = 0; si < 4; ++si) {
#pragma unroll
            for (int r = 0; r < 8; ++r) s[si][r] = 0.0f;
            const unsigned short* base = sKc + (si * 16 + frow) * LDK;
            v16bf k0 = ldfrag(base + grp * 8, base + 16 + grp * 8);
            v16bf k1 = ldfrag(base + 32 + grp * 8, base + 48 + grp * 8);
            s[si] = wmma_bf16(qf[0], k0, s[si]);
            s[si] = wmma_bf16(qf[1], k1, s[si]);
        }

        // ---- online softmax (rows r / r+8 split over lane halves) ----
        float alpha[8], mnew[8];
#pragma unroll
        for (int r = 0; r < 8; ++r) {
            float v = s[0][r] * scale;
#pragma unroll
            for (int si = 1; si < 4; ++si) v = fmaxf(v, s[si][r] * scale);
#pragma unroll
            for (int off = 8; off >= 1; off >>= 1)
                v = fmaxf(v, __shfl_xor(v, off, 16));
            mnew[r] = fmaxf(mrow[r], v);
            alpha[r] = __expf(mrow[r] - mnew[r]);
            mrow[r] = mnew[r];
        }
#pragma unroll
        for (int si = 0; si < 4; ++si)
#pragma unroll
            for (int r = 0; r < 8; ++r)
                s[si][r] = __expf(s[si][r] * scale - mnew[r]);
#pragma unroll
        for (int r = 0; r < 8; ++r) {
            float rs = s[0][r] + s[1][r] + s[2][r] + s[3][r];
#pragma unroll
            for (int off = 8; off >= 1; off >>= 1)
                rs += __shfl_xor(rs, off, 16);
            lsum[r] = lsum[r] * alpha[r] + rs;
#pragma unroll
            for (int dt = 0; dt < 4; ++dt) accO[dt][r] *= alpha[r];
        }

        // ---- P -> bf16 via per-wave LDS, reload as A fragments ----
#pragma unroll
        for (int si = 0; si < 4; ++si)
#pragma unroll
            for (int r = 0; r < 8; ++r)
                sPw[(r + grp * 8) * LDK + si * 16 + frow] = f2bf(s[si][r]);
        LDS_FENCE();
        v16bf pf[2];
        {
            const unsigned short* base = sPw + frow * LDK;
            pf[0] = ldfrag(base + grp * 8, base + 16 + grp * 8);
            pf[1] = ldfrag(base + 32 + grp * 8, base + 48 + grp * 8);
        }

        // ---- O += P V (B operand = Vt rows = columns d of V) ----
#pragma unroll
        for (int dt = 0; dt < 4; ++dt) {
            const unsigned short* base = sVc + (dt * 16 + frow) * LDK;
            v16bf v0 = ldfrag(base + grp * 8, base + 16 + grp * 8);
            v16bf v1 = ldfrag(base + 32 + grp * 8, base + 48 + grp * 8);
            accO[dt] = wmma_bf16(pf[0], v0, accO[dt]);
            accO[dt] = wmma_bf16(pf[1], v1, accO[dt]);
        }
    }

    // ---- epilogue: O / l, f32 store to [B,T,C] ----
    float inv[8];
#pragma unroll
    for (int r = 0; r < 8; ++r) inv[r] = 1.0f / lsum[r];
    const int b = bh >> 4, h = bh & 15;
#pragma unroll
    for (int dt = 0; dt < 4; ++dt)
#pragma unroll
        for (int r = 0; r < 8; ++r) {
            const int t = qt * 128 + wave * 16 + r + grp * 8;
            out[((size_t)b * 2048 + t) * 1024 + h * 64 + dt * 16 + frow] =
                accO[dt][r] * inv[r];
        }
}

// ---------------------------------------------------------------------------
extern "C" void kernel_launch(void* const* d_in, const int* in_sizes, int n_in,
                              void* d_out, int out_size, void* d_ws,
                              size_t ws_size, hipStream_t stream) {
    const float* x    = (const float*)d_in[0];
    const float* W    = (const float*)d_in[1];
    const float* bias = (const float*)d_in[2];
    float* out        = (float*)d_out;

    const size_t HEAD_ELEMS = (size_t)2 * 16 * 2048 * 64;  // 4 Mi elems
    unsigned short* qb  = (unsigned short*)d_ws;
    unsigned short* kb  = qb + HEAD_ELEMS;
    unsigned short* vtb = kb + HEAD_ELEMS;

    qkv_gemm_kernel<<<dim3(24, 32), 256, 0, stream>>>(x, W, bias, qb, kb, vtb);
    mhsa_flash_kernel<<<dim3(32, 16), 256, 0, stream>>>(qb, kb, vtb, out);
}
